// MHAttention_10823317586715
// MI455X (gfx1250) — matrix operand order repaired and checked by reading, not compiled, not run on hardware
//
#include <hip/hip_runtime.h>

// ---------------------------------------------------------------------------
// MHA for MI455X (gfx1250): all GEMMs via v_wmma_f32_16x16x32_bf16 (wave32).
// B=2, T=2048, D=1024, H=16, HD=64.
// Round 5: (a) hardware bf16 converts via __builtin_convertvector (replaces
// ~7-VALU manual RNE per packed dword); (b) batched K/V fragment loads so
// score/PV WMMAs issue back-to-back after one wait; (c) A-fragments loaded
// as 16-B vectors (K-pairs are contiguous within a 16-wide group); LDS P
// stride 72 keeps b128 reads 16-B aligned.
// ---------------------------------------------------------------------------

typedef __attribute__((ext_vector_type(16))) __bf16 v16bf;
typedef __attribute__((ext_vector_type(2)))  __bf16 v2bf;
typedef __attribute__((ext_vector_type(8)))  float  v8f;
typedef __attribute__((ext_vector_type(2)))  float  v2f;

constexpr int Bc = 2, Tc = 2048, Dc = 1024, Hc = 16, HDc = 64;

union AFrag { v16bf v; unsigned u[8]; uint4 q[2]; };
union BFrag { v16bf v; uint4 q[2]; unsigned u[8]; };

__device__ inline unsigned short f2bf(float f) {
  __bf16 h = (__bf16)f;                         // RNE hardware convert
  return __builtin_bit_cast(unsigned short, h);
}
__device__ inline unsigned pack2bf(float a, float b) {
  v2f x; x[0] = a; x[1] = b;
  v2bf t = __builtin_convertvector(x, v2bf);    // v_cvt_pk_bf16_f32 if avail
  return __builtin_bit_cast(unsigned, t);
}
__device__ inline v8f wmma_bf16(v16bf a, v16bf b, v8f c) {
  return __builtin_amdgcn_wmma_f32_16x16x32_bf16(false, a, false, b, (short)0, c,
                                                 false, false);
}
__device__ inline float redmax16(float x) {
#pragma unroll
  for (int m = 1; m < 16; m <<= 1) x = fmaxf(x, __shfl_xor(x, m, 32));
  return x;
}
__device__ inline v8f vzero() {
  return (v8f){0.f, 0.f, 0.f, 0.f, 0.f, 0.f, 0.f, 0.f};
}

// A-fragment (16x32 bf16, ISA 7.12.2): lane m = row; half-wave g selects K
// sub-ranges; K-pairs are contiguous per 16-wide group -> 16-B vector loads.
template <bool ABF16>
__device__ inline AFrag load_afrag(const void* A, int row, int k0, int g) {
  AFrag a;
  if (ABF16) {
    const unsigned short* p =
        (const unsigned short*)A + (long)row * Dc + k0 + (g << 4);
    a.q[0] = *(const uint4*)p;                  // K = g*16 + 0..7
    a.q[1] = *(const uint4*)(p + 8);            // K = g*16 + 8..15
  } else {
    const float* p = (const float*)A + (long)row * Dc + k0 + (g << 4);
    float4 x0 = ((const float4*)p)[0];          // K = g*16 + 0..3
    float4 x1 = ((const float4*)p)[1];          // K = g*16 + 4..7
    float4 x2 = ((const float4*)(p + 8))[0];    // K = g*16 + 8..11
    float4 x3 = ((const float4*)(p + 8))[1];    // K = g*16 + 12..15
    a.u[0] = pack2bf(x0.x, x0.y); a.u[1] = pack2bf(x0.z, x0.w);
    a.u[2] = pack2bf(x1.x, x1.y); a.u[3] = pack2bf(x1.z, x1.w);
    a.u[4] = pack2bf(x2.x, x2.y); a.u[5] = pack2bf(x2.z, x2.w);
    a.u[6] = pack2bf(x3.x, x3.y); a.u[7] = pack2bf(x3.z, x3.w);
  }
  return a;
}
__device__ inline BFrag load_bfrag_w(const float* W, int n, int k0, int g) {
  BFrag b;
  const float* wr = W + (long)n * Dc + k0 + (g << 4);
  float4 w0 = ((const float4*)wr)[0];
  float4 w1 = ((const float4*)wr)[1];
  float4 w2 = ((const float4*)wr)[2];
  float4 w3 = ((const float4*)wr)[3];
  b.u[0] = pack2bf(w0.x, w0.y); b.u[1] = pack2bf(w0.z, w0.w);
  b.u[2] = pack2bf(w1.x, w1.y); b.u[3] = pack2bf(w1.z, w1.w);
  b.u[4] = pack2bf(w2.x, w2.y); b.u[5] = pack2bf(w2.z, w2.w);
  b.u[6] = pack2bf(w3.x, w3.y); b.u[7] = pack2bf(w3.z, w3.w);
  return b;
}
__device__ inline BFrag load_bfrag_bf(const unsigned short* p) {
  BFrag b;
  b.q[0] = *(const uint4*)p;
  b.q[1] = *(const uint4*)(p + 8);
  return b;
}

// ---------------------------------------------------------------------------
// GEMM: Y = A(4096xD) x W(DxD row-major, used transposed).
// One wave = 32x32 output tile (4 independent WMMAs per K-step); next K-slab
// warmed with global_prefetch.
// mode 0: Y->bf16 qp [B,H,T,HD] (scaled)   mode 1: Y->bf16 kp [B,H,T,HD]
// mode 2: Y->bf16 vT [B,H,HD,T]            mode 3: Y->f32 out [M,D]
// ---------------------------------------------------------------------------
template <bool ABF16>
__global__ __launch_bounds__(128) void gemm32(const void* __restrict__ Aptr,
                                              const float* __restrict__ W,
                                              void* __restrict__ Out,
                                              int mode, float scale) {
  const int wave = threadIdx.x >> 5, lane = threadIdx.x & 31;
  const int lane15 = lane & 15, g = lane >> 4;
  const int tilesN = Dc >> 5;                      // 32 tiles of 32 cols
  const int tile = blockIdx.x * 4 + wave;          // 0..4095
  const int tm = (tile / tilesN) << 5;
  const int tn = (tile % tilesN) << 5;

  v8f c00 = vzero(), c01 = vzero(), c10 = vzero(), c11 = vzero();

  for (int k0 = 0; k0 < Dc; k0 += 32) {
    if (k0 + 32 < Dc) {   // warm next K-slab (L2/WGP$) -> global_prefetch_b8
      if (ABF16) {
        __builtin_prefetch((const unsigned short*)Aptr +
                               (long)(tm + lane) * Dc + k0 + 32, 0, 0);
      } else {
        __builtin_prefetch((const float*)Aptr + (long)(tm + lane) * Dc + k0 + 32,
                           0, 0);
      }
      __builtin_prefetch(W + (long)(tn + lane) * Dc + k0 + 32, 0, 0);
    }
    AFrag a0 = load_afrag<ABF16>(Aptr, tm + lane15, k0, g);
    AFrag a1 = load_afrag<ABF16>(Aptr, tm + 16 + lane15, k0, g);
    BFrag b0 = load_bfrag_w(W, tn + lane15, k0, g);
    BFrag b1 = load_bfrag_w(W, tn + 16 + lane15, k0, g);
    c00 = wmma_bf16(a0.v, b0.v, c00);
    c01 = wmma_bf16(a0.v, b1.v, c01);
    c10 = wmma_bf16(a1.v, b0.v, c10);
    c11 = wmma_bf16(a1.v, b1.v, c11);
  }

  v8f ctab[4] = {c00, c01, c10, c11};
#pragma unroll
  for (int mi = 0; mi < 2; ++mi) {
#pragma unroll
    for (int ni = 0; ni < 2; ++ni) {
      const v8f cc = ctab[mi * 2 + ni];
      const int col = tn + ni * 16 + lane15;
#pragma unroll
      for (int r = 0; r < 8; ++r) {
        const int m = tm + mi * 16 + (g << 3) + r;  // output row
        const float val = cc[r] * scale;
        if (mode == 3) {
          ((float*)Out)[(long)m * Dc + col] = val;
        } else {
          const int bb = m >> 11, t = m & (Tc - 1);
          const int hh = col >> 6, hd = col & 63;
          unsigned short* O = (unsigned short*)Out;
          long idx;
          if (mode == 2)  // V transposed: [b,h,hd,t]
            idx = ((long)(bb * Hc + hh) * HDc + hd) * Tc + t;
          else            // qp / kp: [b,h,t,hd]
            idx = ((long)(bb * Hc + hh) * Tc + t) * HDc + hd;
          O[idx] = f2bf(val);
        }
      }
    }
  }
}

// ---------------------------------------------------------------------------
// Flash attention, reference-exact softmax: running max over ALL raw scores
// (pre-mask), causal mask applied to exp() only, denominator = l + 1.
// One wave = 16 queries; 4 waves/block; 64 keys per j-step -> one bpermute
// max tree / alpha-exp / rescale per 64 keys. Denominator l is a 5th WMMA
// accumulator vs an all-ones B fragment.
// ---------------------------------------------------------------------------
__global__ __launch_bounds__(128) void attn_fa(const unsigned short* __restrict__ qp,
                                               const unsigned short* __restrict__ kp,
                                               const unsigned short* __restrict__ vT,
                                               unsigned short* __restrict__ attn) {
  __shared__ __align__(16) unsigned short pbuf[4][16 * 72];  // 16 x 64 (+pad)
  const int wave = threadIdx.x >> 5, lane = threadIdx.x & 31;
  const int lane15 = lane & 15, g = lane >> 4;
  const int bh = blockIdx.x >> 5;                  // (b*H + h)
  const int qbase = (blockIdx.x & 31) * 64 + wave * 16;
  const int b = bh >> 4, h = bh & 15;
  unsigned short* pb = pbuf[wave];

  // Q fragments for the two 32-wide d-chunks of HD=64 (scale pre-folded).
  AFrag aq[2];
#pragma unroll
  for (int cdim = 0; cdim < 2; ++cdim)
    aq[cdim] = load_afrag<true>(qp, bh * Tc + qbase + lane15, cdim * 32, g);

  // All-ones B fragment: row-sum of P via WMMA (result replicated per lane).
  BFrag bones;
#pragma unroll
  for (int p = 0; p < 8; ++p) bones.u[p] = 0x3F803F80u;  // {1.0bf16, 1.0bf16}

  v8f oacc[5];                                     // [0..3]=O chunks, [4]=l
  float mrow[8];
#pragma unroll
  for (int n = 0; n < 5; ++n) oacc[n] = vzero();
#pragma unroll
  for (int r = 0; r < 8; ++r) mrow[r] = -1e30f;

  const unsigned short* kbase = kp + (long)bh * Tc * HDc;
  const unsigned short* vbase = vT + (long)bh * HDc * Tc;

  for (int j0 = 0; j0 < Tc; j0 += 64) {
    if (j0 + 64 < Tc) {   // warm next K/V tiles -> global_prefetch_b8
      __builtin_prefetch(kbase + (long)(j0 + 64 + lane) * HDc, 0, 0);
      __builtin_prefetch(kbase + (long)(j0 + 96 + lane) * HDc, 0, 0);
      __builtin_prefetch(vbase + (long)(lane * 2 + g) * Tc + j0 + 64, 0, 0);
    }
    // S = Q K^T : four 16(q)x16(k) tiles covering 64 keys.  Batch the four
    // B fragments per d-chunk so the WMMAs issue back-to-back after one wait.
    v8f s[4];
    {
      BFrag bk[4];
#pragma unroll
      for (int t = 0; t < 4; ++t)
        bk[t] = load_bfrag_bf(kbase + (long)(j0 + t * 16 + lane15) * HDc + (g << 4));
#pragma unroll
      for (int t = 0; t < 4; ++t) s[t] = wmma_bf16(aq[0].v, bk[t].v, vzero());
#pragma unroll
      for (int t = 0; t < 4; ++t)
        bk[t] = load_bfrag_bf(kbase + (long)(j0 + t * 16 + lane15) * HDc + 32 +
                              (g << 4));
#pragma unroll
      for (int t = 0; t < 4; ++t) s[t] = wmma_bf16(aq[1].v, bk[t].v, s[t]);
    }

    // ONE online-max update per 64 keys (raw scores, pre-mask).
    float alpha[8];
#pragma unroll
    for (int r = 0; r < 8; ++r) {
      const float x = redmax16(fmaxf(fmaxf(s[0][r], s[1][r]),
                                     fmaxf(s[2][r], s[3][r])));
      const float mn = fmaxf(mrow[r], x);
      alpha[r] = __expf(mrow[r] - mn);
      mrow[r] = mn;
    }
    // Masked exp, P -> LDS (bf16, A-layout reload below).
#pragma unroll
    for (int r = 0; r < 8; ++r) {
      const int i = qbase + (g << 3) + r;          // query index
      const int row = (g << 3) + r;
#pragma unroll
      for (int t = 0; t < 4; ++t) {
        const int j = j0 + t * 16 + lane15;
        const float pv = (j <= i) ? __expf(s[t][r] - mrow[r]) : 0.f;
        pb[row * 72 + t * 16 + lane15] = f2bf(pv);
      }
    }
#pragma unroll
    for (int n = 0; n < 5; ++n)
#pragma unroll
      for (int r = 0; r < 8; ++r) oacc[n][r] *= alpha[r];

    asm volatile("s_wait_dscnt 0x0" ::: "memory");

    // Reload P as two bf16 A fragments (16 q x 64 keys), 16-B aligned.
    AFrag ap0, ap1;
    {
      const unsigned short* pr = pb + lane15 * 72 + (g << 4);
      ap0.q[0] = *(const uint4*)pr;
      ap0.q[1] = *(const uint4*)(pr + 8);
      ap1.q[0] = *(const uint4*)(pr + 32);
      ap1.q[1] = *(const uint4*)(pr + 40);
    }

    // O += P*V (4 chunks x 2 key-halves); l += P*ones.  Batched V loads.
    {
      BFrag bv[4];
#pragma unroll
      for (int n = 0; n < 4; ++n)
        bv[n] = load_bfrag_bf(vbase + (long)(n * 16 + lane15) * Tc + j0 + (g << 4));
#pragma unroll
      for (int n = 0; n < 4; ++n) oacc[n] = wmma_bf16(ap0.v, bv[n].v, oacc[n]);
      oacc[4] = wmma_bf16(ap0.v, bones.v, oacc[4]);
#pragma unroll
      for (int n = 0; n < 4; ++n)
        bv[n] = load_bfrag_bf(vbase + (long)(n * 16 + lane15) * Tc + j0 + 32 +
                              (g << 4));
#pragma unroll
      for (int n = 0; n < 4; ++n) oacc[n] = wmma_bf16(ap1.v, bv[n].v, oacc[n]);
      oacc[4] = wmma_bf16(ap1.v, bones.v, oacc[4]);
    }
  }

  // out = O / (l + 1), written bf16 as [b, t, h*64+d] for the final GEMM.
#pragma unroll
  for (int n = 0; n < 4; ++n) {
    const int col = h * 64 + n * 16 + lane15;
#pragma unroll
    for (int r = 0; r < 8; ++r) {
      const int t = qbase + (g << 3) + r;
      const float val = oacc[n][r] / (oacc[4][r] + 1.0f);
      attn[((long)(b * Tc + t)) * Dc + col] = f2bf(val);
    }
  }
}

// ---------------------------------------------------------------------------
extern "C" void kernel_launch(void* const* d_in, const int* in_sizes, int n_in,
                              void* d_out, int out_size, void* d_ws, size_t ws_size,
                              hipStream_t stream) {
  const float* q  = (const float*)d_in[0];
  const float* k  = (const float*)d_in[1];
  const float* v  = (const float*)d_in[2];
  const float* Wq = (const float*)d_in[3];
  const float* Wk = (const float*)d_in[4];
  const float* Wv = (const float*)d_in[5];
  const float* Wo = (const float*)d_in[6];

  const size_t seg = (size_t)Bc * Tc * Dc * sizeof(unsigned short);  // 8 MB
  char* ws = (char*)d_ws;
  unsigned short* qp   = (unsigned short*)(ws);
  unsigned short* kp   = (unsigned short*)(ws + seg);
  unsigned short* vT   = (unsigned short*)(ws + 2 * seg);
  unsigned short* attn = (unsigned short*)(ws + 3 * seg);

  // scores scale = log(B)/sqrt(HD) folded into Q projection.
  const float qscale = 0.69314718055994531f / 8.0f;

  const dim3 ggrid(1024), gblk(128);                // 4096 32x32 wave-tiles
  hipLaunchKernelGGL((gemm32<false>), ggrid, gblk, 0, stream, q, Wq, qp, 0, qscale);
  hipLaunchKernelGGL((gemm32<false>), ggrid, gblk, 0, stream, k, Wk, kp, 1, 1.0f);
  hipLaunchKernelGGL((gemm32<false>), ggrid, gblk, 0, stream, v, Wv, vT, 2, 1.0f);
  hipLaunchKernelGGL(attn_fa, dim3(1024), gblk, 0, stream, qp, kp, vT, attn);
  hipLaunchKernelGGL((gemm32<true>), ggrid, gblk, 0, stream, attn, Wo, d_out, 3, 1.0f);
}
